// SpatialAttention_68083821576702
// MI455X (gfx1250) — matrix operand order repaired and checked by reading, not compiled, not run on hardware
//
#include <hip/hip_runtime.h>

// SpatialAttention on MI455X (gfx1250): flash-attention formulation.
// scores[n,s,t] = sum_c k[n,c,s]*q[n,c,t]/sqrt(c); softmax over s; out = V.P.
// Online softmax over s-blocks; matmuls via v_wmma_f32_16x16x32_bf16.
// Softmax runs in the log2 domain (log2e folded into the Q scale) so the
// exp path is bare v_exp_f32 (__builtin_amdgcn_exp2f) with no multiplies.
// LDS layouts chosen so every WMMA fragment = two aligned ds_load_b128.

typedef __attribute__((ext_vector_type(16))) __bf16 v16bf;
typedef __attribute__((ext_vector_type(8)))  __bf16 v8bf;
typedef __attribute__((ext_vector_type(8)))  float  v8f;

#define NB    4     // batch
#define CD    128   // channels (features per token)
#define HWN   4096  // tokens (h*w)
#define TT    64    // query-token tile per workgroup
#define SB    64    // key-token block per iteration
#define NTHR  128   // 4 waves of 32
#define CROW  (CD + 8)   // 136 elems = 272 B, multiple of 16 B
#define SROW  (SB + 8)   // 72 elems  = 144 B, multiple of 16 B

// Build a 16-element bf16 fragment from two aligned 16-byte LDS loads.
__device__ __forceinline__ v16bf ld_frag(const __bf16* p0, const __bf16* p1) {
    v8bf a = *reinterpret_cast<const v8bf*>(p0);
    v8bf b = *reinterpret_cast<const v8bf*>(p1);
    return __builtin_shufflevector(a, b, 0, 1, 2, 3, 4, 5, 6, 7,
                                         8, 9, 10, 11, 12, 13, 14, 15);
}

// Pack two f32 -> two bf16 in one dword.
__device__ __forceinline__ unsigned pk2(float a, float b) {
    unsigned short ua = __builtin_bit_cast(unsigned short, (__bf16)a);
    unsigned short ub = __builtin_bit_cast(unsigned short, (__bf16)b);
    return ((unsigned)ub << 16) | ua;
}

__global__ __launch_bounds__(NTHR)
void spatial_attn_flash_bf16(const float* __restrict__ key,
                             const float* __restrict__ query,
                             const float* __restrict__ value,
                             float* __restrict__ out) {
    // qt/kt staging space is dead after the main loop; reuse it as the f32
    // output-transpose buffer so final global stores are coalesced along t.
    __shared__ union SmemA {
        struct { __bf16 qt[TT][CROW]; __bf16 kt[SB][CROW]; } s; // 34816 B
        float ot[CD][TT + 4];                                   // 34816 B
    } smA;
    __shared__ __bf16 vs[CD][SROW];                 // V block [e][s], 18432 B
    __shared__ __bf16 ps[NTHR / 32][16][SROW];      // per-wave P [t][s], 9216 B

    const int tid  = threadIdx.x;
    const int wave = tid >> 5;
    const int lane = tid & 31;
    const int half = lane >> 4;   // 0: lanes 0-15, 1: lanes 16-31
    const int l16  = lane & 15;
    const int t0l  = wave * 16;   // wave's first local query row

    const int bat   = blockIdx.x / (HWN / TT);
    const int tile  = blockIdx.x % (HWN / TT);
    const int tbase = tile * TT;
    const size_t base = (size_t)bat * CD * HWN;

    // 1/sqrt(128) * log2(e): scores land pre-scaled for exp2-domain softmax.
    const float qscale = 0.08838834764831845f * 1.4426950408889634f;

    // ---- Stage Q tile transposed into LDS: qt[t][c] (bf16, pre-scaled).
    for (int r = 0; r < (CD * TT / 4) / NTHR; ++r) {
        int idx = r * NTHR + tid;       // 0 .. 2047 float4 items
        int c   = idx >> 4;             // TT/4 = 16 float4 per c-row
        int f4  = idx & 15;
        const float4 v = *reinterpret_cast<const float4*>(
            query + base + (size_t)c * HWN + tbase + f4 * 4);
        smA.s.qt[f4 * 4 + 0][c] = (__bf16)(v.x * qscale);
        smA.s.qt[f4 * 4 + 1][c] = (__bf16)(v.y * qscale);
        smA.s.qt[f4 * 4 + 2][c] = (__bf16)(v.z * qscale);
        smA.s.qt[f4 * 4 + 3][c] = (__bf16)(v.w * qscale);
    }
    __syncthreads();

    // ---- Q A-fragments (M=t, K=c), 4 chunks of K=32; each = 2x ds_load_b128.
    v16bf aq[4];
#pragma unroll
    for (int cc = 0; cc < 4; ++cc)
        aq[cc] = ld_frag(&smA.s.qt[t0l + l16][cc * 32 + half * 8],
                         &smA.s.qt[t0l + l16][cc * 32 + 16 + half * 8]);

    // ---- Running softmax state per owned row (row = i + half*8), log2 domain.
    float m[8], l[8];
    v8f   O[8];                      // O tile (M=t, N=e), 8 chunks of e=16
#pragma unroll
    for (int i = 0; i < 8; ++i) { m[i] = -1.0e30f; l[i] = 0.0f; }
#pragma unroll
    for (int ec = 0; ec < 8; ++ec)
#pragma unroll
        for (int i = 0; i < 8; ++i) O[ec][i] = 0.0f;

    // ---- Main loop over key/value blocks.
    for (int sb = 0; sb < HWN / SB; ++sb) {
        const int sbase = sb * SB;

        // Stage K transposed (kt[s][c], scalar scatter) and V natural
        // (vs[e][s], packed uint2 stores).
        for (int r = 0; r < (CD * SB / 4) / NTHR; ++r) {
            int idx = r * NTHR + tid;
            int c   = idx >> 4;
            int f4  = idx & 15;
            const float4 kv = *reinterpret_cast<const float4*>(
                key + base + (size_t)c * HWN + sbase + f4 * 4);
            smA.s.kt[f4 * 4 + 0][c] = (__bf16)kv.x;
            smA.s.kt[f4 * 4 + 1][c] = (__bf16)kv.y;
            smA.s.kt[f4 * 4 + 2][c] = (__bf16)kv.z;
            smA.s.kt[f4 * 4 + 3][c] = (__bf16)kv.w;
            const float4 vv = *reinterpret_cast<const float4*>(
                value + base + (size_t)c * HWN + sbase + f4 * 4);
            uint2 pv = make_uint2(pk2(vv.x, vv.y), pk2(vv.z, vv.w));
            *reinterpret_cast<uint2*>(&vs[c][f4 * 4]) = pv;
        }
        // Prefetch next block (global_prefetch_b8): one row per thread.
        if (sb + 1 < HWN / SB) {
            __builtin_prefetch(key   + base + (size_t)tid * HWN + sbase + SB, 0, 1);
            __builtin_prefetch(value + base + (size_t)tid * HWN + sbase + SB, 0, 1);
        }
        __syncthreads();

        // ---- S = Qt . K : D (M=t, N=s), 4 N-chunks; K=c in 4 steps of 32.
        v8f S[4];
#pragma unroll
        for (int nc = 0; nc < 4; ++nc)
#pragma unroll
            for (int i = 0; i < 8; ++i) S[nc][i] = 0.0f;

#pragma unroll
        for (int nc = 0; nc < 4; ++nc) {
#pragma unroll
            for (int cc = 0; cc < 4; ++cc) {
                // B (K=c, N=s): two runs of 8 c (8h, 16+8h) at fixed s -> kt[s][c...]
                v16bf bk = ld_frag(&smA.s.kt[nc * 16 + l16][cc * 32 + half * 8],
                                   &smA.s.kt[nc * 16 + l16][cc * 32 + 16 + half * 8]);
                S[nc] = __builtin_amdgcn_wmma_f32_16x16x32_bf16(
                    false, aq[cc], false, bk, (short)0, S[nc], false, false);
            }
        }

        // ---- Online softmax over s, log2 domain (bare v_exp_f32).
        float rmax[8];
#pragma unroll
        for (int i = 0; i < 8; ++i)
            rmax[i] = fmaxf(fmaxf(S[0][i], S[1][i]), fmaxf(S[2][i], S[3][i]));
#pragma unroll
        for (int msk = 8; msk >= 1; msk >>= 1)
#pragma unroll
            for (int i = 0; i < 8; ++i)
                rmax[i] = fmaxf(rmax[i], __shfl_xor(rmax[i], msk, 32));

        float alpha[8], rsum[8];
#pragma unroll
        for (int i = 0; i < 8; ++i) {
            float mn = fmaxf(m[i], rmax[i]);
            alpha[i] = __builtin_amdgcn_exp2f(m[i] - mn);
            m[i] = mn;
            rsum[i] = 0.0f;
        }
#pragma unroll
        for (int nc = 0; nc < 4; ++nc)
#pragma unroll
            for (int i = 0; i < 8; ++i) {
                float p = __builtin_amdgcn_exp2f(S[nc][i] - m[i]);
                S[nc][i] = p;
                rsum[i] += p;
            }
#pragma unroll
        for (int msk = 8; msk >= 1; msk >>= 1)
#pragma unroll
            for (int i = 0; i < 8; ++i)
                rsum[i] += __shfl_xor(rsum[i], msk, 32);
#pragma unroll
        for (int i = 0; i < 8; ++i) l[i] = l[i] * alpha[i] + rsum[i];
#pragma unroll
        for (int ec = 0; ec < 8; ++ec)
#pragma unroll
            for (int i = 0; i < 8; ++i) O[ec][i] *= alpha[i];

        // ---- P: C/D layout -> LDS [t][s] (scatter b16; same-wave, in-order).
#pragma unroll
        for (int nc = 0; nc < 4; ++nc)
#pragma unroll
            for (int i = 0; i < 8; ++i)
                ps[wave][i + half * 8][nc * 16 + l16] = (__bf16)S[nc][i];

        // ---- O += P . V : A (M=t, K=s) x B (K=s, N=e); K=s in 2 steps of 32.
#pragma unroll
        for (int kc = 0; kc < 2; ++kc) {
            v16bf ap = ld_frag(&ps[wave][l16][kc * 32 + half * 8],
                               &ps[wave][l16][kc * 32 + 16 + half * 8]);
#pragma unroll
            for (int ec = 0; ec < 8; ++ec) {
                // B (K=s, N=e): two runs of 8 s (8h, 16+8h) at fixed e -> vs[e][s...]
                v16bf bv = ld_frag(&vs[ec * 16 + l16][kc * 32 + half * 8],
                                   &vs[ec * 16 + l16][kc * 32 + 16 + half * 8]);
                O[ec] = __builtin_amdgcn_wmma_f32_16x16x32_bf16(
                    false, ap, false, bv, (short)0, O[ec], false, false);
            }
        }
        __syncthreads(); // protect kt/vs before next block's staging
    }

    // ---- Epilogue: O[t][e] * (1/l[t]) -> LDS ot[e][t] (f32), then coalesced
    //      global stores along t. 8x v_rcp_f32 instead of 64 f32 divides.
    float rl[8];
#pragma unroll
    for (int i = 0; i < 8; ++i) rl[i] = __builtin_amdgcn_rcpf(l[i]);
#pragma unroll
    for (int ec = 0; ec < 8; ++ec) {
        const int e  = ec * 16 + l16;
        const int tl = t0l + half * 8;
        float4 w0, w1;
        w0.x = O[ec][0] * rl[0];  w0.y = O[ec][1] * rl[1];
        w0.z = O[ec][2] * rl[2];  w0.w = O[ec][3] * rl[3];
        w1.x = O[ec][4] * rl[4];  w1.y = O[ec][5] * rl[5];
        w1.z = O[ec][6] * rl[6];  w1.w = O[ec][7] * rl[7];
        *reinterpret_cast<float4*>(&smA.ot[e][tl])     = w0;
        *reinterpret_cast<float4*>(&smA.ot[e][tl + 4]) = w1;
    }
    __syncthreads();
    for (int r = 0; r < (CD * TT / 4) / NTHR; ++r) {
        int idx = r * NTHR + tid;
        int e   = idx >> 4;
        int f4  = idx & 15;
        *reinterpret_cast<float4*>(out + base + (size_t)e * HWN + tbase + f4 * 4) =
            *reinterpret_cast<const float4*>(&smA.ot[e][f4 * 4]);
    }
}

extern "C" void kernel_launch(void* const* d_in, const int* in_sizes, int n_in,
                              void* d_out, int out_size, void* d_ws, size_t ws_size,
                              hipStream_t stream) {
    const float* key   = (const float*)d_in[0];
    const float* query = (const float*)d_in[1];
    const float* value = (const float*)d_in[2];
    float*       out   = (float*)d_out;
    (void)in_sizes; (void)n_in; (void)out_size; (void)d_ws; (void)ws_size;

    dim3 grid(NB * (HWN / TT));   // 4 batches x 64 query tiles = 256 blocks
    dim3 block(NTHR);             // 128 threads = 4 waves (wave32)
    spatial_attn_flash_bf16<<<grid, block, 0, stream>>>(key, query, value, out);
}
